// ScaBlock_23914377904609
// MI455X (gfx1250) — matrix operand order repaired and checked by reading, not compiled, not run on hardware
//
#include <hip/hip_runtime.h>
#include <hip/hip_bf16.h>
#include <math.h>

// ============================================================================
// MI455X (gfx1250) implementation of the DTCWT-SVT + LeFF transformer block.
// All matmuls run on v_wmma_f32_16x16x32_bf16 (f32 accumulate); weight
// matrices are pre-swizzled into WMMA B-fragment order and staged into LDS
// per workgroup via gfx1250 async global->LDS copies (ASYNCcnt).
// Bandwidth-bound workload (~1GB traffic vs ~40 GFLOP) -> streaming passes
// stay f32, GEMM inputs downcast to bf16 (tf32-like).
// Workspace arena requirement: ~833 MB.
// ============================================================================

typedef __attribute__((ext_vector_type(16))) __bf16 v16bf;
typedef __attribute__((ext_vector_type(8)))  __bf16 v8bf;
typedef __attribute__((ext_vector_type(8)))  float  v8f;
typedef __attribute__((ext_vector_type(4)))  int    v4i;

#define DEVINL __device__ __forceinline__

#define BB   256
#define NNP  1024
#define CCH  64
#define HID_ 256
#define RROWS (BB*6*256)    // 393216 subband rows
#define BN   (BB*NNP)       // 262144
#define S1   (BB*CCH*1024)  // elems of one (B,C,32,32) tensor

__constant__ float c_H0[13] = {-0.0017578f, 0.0f, 0.0222656f, -0.046875f,
    -0.0482422f, 0.296875f, 0.5554688f, 0.296875f, -0.0482422f, -0.046875f,
    0.0222656f, 0.0f, -0.0017578f};
__constant__ float c_H1[19] = {-7.06e-05f, 0.0f, 0.0013419f, -0.0018834f,
    -0.0071568f, 0.023856f, 0.0556431f, -0.0516881f, -0.2997576f, 0.5594308f,
    -0.2997576f, -0.0516881f, 0.0556431f, 0.023856f, -0.0071568f, -0.0018834f,
    0.0013419f, 0.0f, -7.06e-05f};

#define ISQ2F 0.70710678118654752f

DEVINL int mir32(int j) { return j < 0 ? -1 - j : (j > 31 ? 63 - j : j); }

DEVINL __bf16 f2bf(float f) {
  unsigned u = __float_as_uint(f);
  unsigned r = u + 0x7FFFu + ((u >> 16) & 1u);   // round-to-nearest-even
  unsigned short s = (unsigned short)(r >> 16);
  return __builtin_bit_cast(__bf16, s);
}

DEVINL float geluf(float x) { return 0.5f * x * (1.0f + erff(x * ISQ2F)); }

DEVINL float wred32(float v) {
  v += __shfl_xor(v, 16, 32);
  v += __shfl_xor(v, 8, 32);
  v += __shfl_xor(v, 4, 32);
  v += __shfl_xor(v, 2, 32);
  v += __shfl_xor(v, 1, 32);
  return v;
}

// inverse-filter taps: G0 (19) from H1, G1 (13) from H0, sign (-1)^(n+1)
DEVINL float g0tap(int k) { return (k & 1) ? c_H1[k] : -c_H1[k]; }
DEVINL float g1tap(int k) { return (k & 1) ? c_H0[k] : -c_H0[k]; }

// ---------------------------------------------------------------------------
// gfx1250 async global->LDS copy (16B per lane, tracked by ASYNCcnt), with
// a plain VGPR-bounce fallback if the builtin is unavailable.
// Builtin signature (from hipcc diagnostic): (v4i AS1*, v4i AS3*, Imm, Imm).
// ---------------------------------------------------------------------------
DEVINL void copy16_g2l(const __bf16* __restrict__ g, __bf16* l) {
#if __has_builtin(__builtin_amdgcn_global_load_async_to_lds_b128)
  __builtin_amdgcn_global_load_async_to_lds_b128(
      (__attribute__((address_space(1))) v4i*)(__bf16*)g,
      (__attribute__((address_space(3))) v4i*)l,
      0, 0);
#else
  *(v8bf*)l = *(const v8bf*)g;
#endif
}

DEVINL void async_wait0() {
#if __has_builtin(__builtin_amdgcn_global_load_async_to_lds_b128)
#if __has_builtin(__builtin_amdgcn_s_wait_asynccnt)
  __builtin_amdgcn_s_wait_asynccnt(0);
#else
  asm volatile("s_wait_asynccnt 0" ::: "memory");
#endif
#endif
}

// ---------------------------------------------------------------------------
// LN1: (B,N,C) f32 -> normalized, transposed to NCHW f32.  One wave per row.
// ---------------------------------------------------------------------------
__global__ void __launch_bounds__(128) ln1_to_nchw(
    const float* __restrict__ x, const float* __restrict__ g,
    const float* __restrict__ b, float* __restrict__ xi) {
  const int lane = threadIdx.x & 31;
  const size_t r = (size_t)blockIdx.x * 4 + (threadIdx.x >> 5);
  const float* row = x + r * CCH;
  float v0 = row[lane], v1 = row[lane + 32];
  float m = wred32(v0 + v1) * (1.0f / 64.0f);
  float d0 = v0 - m, d1 = v1 - m;
  float var = wred32(d0 * d0 + d1 * d1) * (1.0f / 64.0f);
  float inv = rsqrtf(var + 1e-5f);
  int bb = (int)(r >> 10), hw = (int)(r & 1023);
  size_t base = (size_t)bb * CCH * 1024;
  xi[base + (size_t)lane * 1024 + hw]        = d0 * inv * g[lane] + b[lane];
  xi[base + (size_t)(lane + 32) * 1024 + hw] = d1 * inv * g[lane + 32] + b[lane + 32];
}

// LN2: (B,N,C) f32 -> bf16 row-major (BN,64)
__global__ void __launch_bounds__(128) ln2_to_bf16(
    const float* __restrict__ x, const float* __restrict__ g,
    const float* __restrict__ b, __bf16* __restrict__ out) {
  const int lane = threadIdx.x & 31;
  const size_t r = (size_t)blockIdx.x * 4 + (threadIdx.x >> 5);
  const float* row = x + r * CCH;
  float v0 = row[lane], v1 = row[lane + 32];
  float m = wred32(v0 + v1) * (1.0f / 64.0f);
  float d0 = v0 - m, d1 = v1 - m;
  float var = wred32(d0 * d0 + d1 * d1) * (1.0f / 64.0f);
  float inv = rsqrtf(var + 1e-5f);
  out[r * CCH + lane]      = f2bf(d0 * inv * g[lane] + b[lane]);
  out[r * CCH + lane + 32] = f2bf(d1 * inv * g[lane + 32] + b[lane + 32]);
}

// ---------------------------------------------------------------------------
// Forward DTCWT: separable filters with symmetric padding, one 32x32 plane
// per workgroup, staged in LDS.
// ---------------------------------------------------------------------------
__global__ void __launch_bounds__(1024) fwd_row_filter(
    const float* __restrict__ xi, float* __restrict__ lo, float* __restrict__ hi) {
  __shared__ float t[32][33];
  const int tx = threadIdx.x, ty = threadIdx.y;
  const size_t base = (size_t)blockIdx.x * 1024;
  t[ty][tx] = xi[base + ty * 32 + tx];
  __syncthreads();
  float aL = 0.f, aH = 0.f;
#pragma unroll
  for (int k = 0; k < 19; ++k) aH += c_H1[k] * t[ty][mir32(tx + k - 9)];
#pragma unroll
  for (int k = 0; k < 13; ++k) aL += c_H0[k] * t[ty][mir32(tx + k - 6)];
  lo[base + ty * 32 + tx] = aL;
  hi[base + ty * 32 + tx] = aH;
}

__global__ void __launch_bounds__(1024) fwd_col_filter(
    const float* __restrict__ lo, const float* __restrict__ hi,
    float* __restrict__ ll, float* __restrict__ lh,
    float* __restrict__ hl, float* __restrict__ hh) {
  __shared__ float tl[32][33], th[32][33];
  const int tx = threadIdx.x, ty = threadIdx.y;
  const size_t base = (size_t)blockIdx.x * 1024;
  tl[ty][tx] = lo[base + ty * 32 + tx];
  th[ty][tx] = hi[base + ty * 32 + tx];
  __syncthreads();
  float aLL = 0.f, aLH = 0.f, aHL = 0.f, aHH = 0.f;
#pragma unroll
  for (int k = 0; k < 13; ++k) {
    int j = mir32(ty + k - 6); float h0 = c_H0[k];
    aLL += h0 * tl[j][tx]; aHL += h0 * th[j][tx];
  }
#pragma unroll
  for (int k = 0; k < 19; ++k) {
    int j = mir32(ty + k - 9); float h1 = c_H1[k];
    aLH += h1 * tl[j][tx]; aHH += h1 * th[j][tx];
  }
  size_t o = base + ty * 32 + tx;
  ll[o] = aLL; lh[o] = aLH; hl[o] = aHL; hh[o] = aHH;
}

__global__ void __launch_bounds__(256) ll_scale(
    float* __restrict__ ll, const float* __restrict__ wll) {
  size_t i = (size_t)blockIdx.x * blockDim.x + threadIdx.x;
  ll[i] *= wll[i & (CCH * 1024 - 1)];
}

// ---------------------------------------------------------------------------
// q2c: interleaved quads -> 6 complex subbands, packed bf16 (R x 128)
// cols 0..63 = real (c), 64..127 = imag (c).  Row r = ((b*6+o)*16+i)*16+j.
// ---------------------------------------------------------------------------
__global__ void __launch_bounds__(64) q2c_pack(
    const float* __restrict__ lh, const float* __restrict__ hl,
    const float* __restrict__ hh, __bf16* __restrict__ XRI) {
  const int c = threadIdx.x;
  const int bij = blockIdx.x;
  const int b = bij >> 8, ij = bij & 255;
  const int i = ij >> 4, j = ij & 15;
  const size_t plane = ((size_t)b * CCH + c) * 1024;
  const int p00 = (2 * i) * 32 + 2 * j;
  auto doSrc = [&](const float* src, int oA, int oB) {
    float a  = src[plane + p00]      * ISQ2F;
    float bq = src[plane + p00 + 1]  * ISQ2F;
    float cq = src[plane + p00 + 32] * ISQ2F;
    float d  = src[plane + p00 + 33] * ISQ2F;
    size_t rA = (((size_t)b * 6 + oA) * 256 + ij) * 128;
    size_t rB = (((size_t)b * 6 + oB) * 256 + ij) * 128;
    XRI[rA + c]      = f2bf(a - d);  XRI[rA + 64 + c] = f2bf(bq + cq);
    XRI[rB + c]      = f2bf(a + d);  XRI[rB + 64 + c] = f2bf(bq - cq);
  };
  doSrc(lh, 0, 5);
  doSrc(hh, 1, 4);
  doSrc(hl, 2, 3);
}

// c2q: subbands f32 (R x 128) -> interleaved quads in three planes
__global__ void __launch_bounds__(64) c2q_unpack(
    const float* __restrict__ H2, float* __restrict__ lh2,
    float* __restrict__ hl2, float* __restrict__ hh2) {
  const int c = threadIdx.x;
  const int bij = blockIdx.x;
  const int b = bij >> 8, ij = bij & 255;
  const int i = ij >> 4, j = ij & 15;
  const size_t plane = ((size_t)b * CCH + c) * 1024;
  const int p00 = (2 * i) * 32 + 2 * j;
  auto doT = [&](float* dst, int oA, int oB) {
    size_t rA = (((size_t)b * 6 + oA) * 256 + ij) * 128;
    size_t rB = (((size_t)b * 6 + oB) * 256 + ij) * 128;
    float w1r = H2[rA + c], w1i = H2[rA + 64 + c];
    float w2r = H2[rB + c], w2i = H2[rB + 64 + c];
    dst[plane + p00]      = (w1r + w2r) * ISQ2F;
    dst[plane + p00 + 1]  = (w1i + w2i) * ISQ2F;
    dst[plane + p00 + 32] = (w1i - w2i) * ISQ2F;
    dst[plane + p00 + 33] = (w2r - w1r) * ISQ2F;
  };
  doT(lh2, 0, 5);
  doT(hh2, 1, 4);
  doT(hl2, 2, 3);
}

// ---------------------------------------------------------------------------
// Inverse DTCWT filters + residual + transpose back to (B,N,C).
// ---------------------------------------------------------------------------
__global__ void __launch_bounds__(1024) inv_col_filter(
    const float* __restrict__ ll, const float* __restrict__ lh2,
    const float* __restrict__ hl2, const float* __restrict__ hh2,
    float* __restrict__ lo2, float* __restrict__ hi2) {
  __shared__ float tll[32][33], tlh[32][33], thl[32][33], thh[32][33];
  const int tx = threadIdx.x, ty = threadIdx.y;
  const size_t base = (size_t)blockIdx.x * 1024;
  tll[ty][tx] = ll[base + ty * 32 + tx];
  tlh[ty][tx] = lh2[base + ty * 32 + tx];
  thl[ty][tx] = hl2[base + ty * 32 + tx];
  thh[ty][tx] = hh2[base + ty * 32 + tx];
  __syncthreads();
  float aL = 0.f, aH = 0.f;
#pragma unroll
  for (int k = 0; k < 19; ++k) {
    int j = mir32(ty + k - 9); float gg = g0tap(k);
    aL += gg * tll[j][tx]; aH += gg * thl[j][tx];
  }
#pragma unroll
  for (int k = 0; k < 13; ++k) {
    int j = mir32(ty + k - 6); float gg = g1tap(k);
    aL += gg * tlh[j][tx]; aH += gg * thh[j][tx];
  }
  lo2[base + ty * 32 + tx] = aL;
  hi2[base + ty * 32 + tx] = aH;
}

__global__ void __launch_bounds__(1024) inv_row_resid(
    const float* __restrict__ lo2, const float* __restrict__ hi2,
    const float* __restrict__ x, float* __restrict__ y1) {
  __shared__ float tlo[32][33], thi[32][33];
  const int tx = threadIdx.x, ty = threadIdx.y;
  const size_t p = blockIdx.x;
  const size_t base = p * 1024;
  tlo[ty][tx] = lo2[base + ty * 32 + tx];
  thi[ty][tx] = hi2[base + ty * 32 + tx];
  __syncthreads();
  float y = 0.f;
#pragma unroll
  for (int k = 0; k < 19; ++k) y += g0tap(k) * tlo[ty][mir32(tx + k - 9)];
#pragma unroll
  for (int k = 0; k < 13; ++k) y += g1tap(k) * thi[ty][mir32(tx + k - 6)];
  int b = (int)(p >> 6), c = (int)(p & 63);
  size_t idx = ((size_t)b * 1024 + ty * 32 + tx) * CCH + c;
  y1[idx] = x[idx] + y;
}

// ---------------------------------------------------------------------------
// Weight prep: fold 4-block complex MLP into dense 128x128 real matrices,
// then swizzle any K x N f32 matrix into WMMA B-fragment order (bf16):
// element (nt,kt,lane,t) = W[kt*32 + (lane>>4)*16 + t][nt*16 + (lane&15)].
// ---------------------------------------------------------------------------
__global__ void __launch_bounds__(256) build_wbig(
    const float* __restrict__ w, const float* __restrict__ bvec,
    float* __restrict__ Wbig, float* __restrict__ bbig) {
  int gid = blockIdx.x * blockDim.x + threadIdx.x;  // 128*128
  int col = gid & 127, row = gid >> 7;
  int cs = col >> 6, c2 = col & 63, nbc = c2 >> 4, k = c2 & 15;
  int rs = row >> 6, r2 = row & 63, nbr = r2 >> 4, d = r2 & 15;
  float v = 0.f;
  if (nbr == nbc) {
    float wr = w[((0 * 4 + nbc) * 16 + d) * 16 + k];
    float wi = w[((1 * 4 + nbc) * 16 + d) * 16 + k];
    v = (cs == 0) ? ((rs == 0) ? wr : -wi) : ((rs == 0) ? wi : wr);
  }
  Wbig[row * 128 + col] = v;
  if (row == 0) bbig[col] = bvec[cs * 64 + nbc * 16 + k];
}

__global__ void __launch_bounds__(256) prep_wfrag(
    const float* __restrict__ Wsrc, __bf16* __restrict__ Wfrag, int K, int N) {
  int gid = blockIdx.x * blockDim.x + threadIdx.x;
  if (gid >= K * N) return;
  int t = gid & 15, lane = (gid >> 4) & 31, rest = gid >> 9;
  int KT = K >> 5;
  int kt = rest % KT, nt = rest / KT;
  int n = nt * 16 + (lane & 15);
  int k = kt * 32 + ((lane >> 4) << 4) + t;
  Wfrag[gid] = f2bf(Wsrc[k * N + n]);
}

// ---------------------------------------------------------------------------
// WMMA GEMM: Out = act(A[M,K]bf16 @ W[K,N]bf16 + bias).
// The entire pre-swizzled weight (32KB for all shapes used here) is staged
// into LDS once per block with async global->LDS copies; every wave then
// feeds its B fragments from ds_load_b128 while A streams from HBM.
// One wave = 16 rows, all N columns held in v8f accumulators (NT per wave).
// ACT: 0 none, 1 relu, 2 gelu.  STORE: 0 f32 row, 1 bf16 row, 2 f32 NCHW,
// 3 f32 row + residual.
// ---------------------------------------------------------------------------
template <int K, int N, int ACT, int STORE>
__global__ void __launch_bounds__(256) gemm_bf16_wmma(
    const __bf16* __restrict__ A, const __bf16* __restrict__ Wfrag,
    const float* __restrict__ bias, void* __restrict__ Out,
    const float* __restrict__ resid) {
  constexpr int KT = K / 32, NT = N / 16;
  __shared__ __bf16 sW[K * N];              // 32 KB

  const int tid = threadIdx.x;
  const int lane = tid & 31;
  const int mtile = blockIdx.x * (blockDim.x >> 5) + (tid >> 5);
  const int khalf = lane >> 4;

  // ---- stage weights: 256 threads x 8 chunks x 16B = K*N*2 bytes ----------
  constexpr int CHUNKS = (K * N) / (256 * 8);  // 8 for all instantiations
#pragma unroll
  for (int it = 0; it < CHUNKS; ++it) {
    int e = (it * 256 + tid) * 8;            // element offset (8 bf16 = 16B)
    copy16_g2l(Wfrag + e, &sW[e]);
  }
  async_wait0();
  __syncthreads();

  const v8f vzero = {0.f, 0.f, 0.f, 0.f, 0.f, 0.f, 0.f, 0.f};
  v8f acc[NT];
#pragma unroll
  for (int nt = 0; nt < NT; ++nt) acc[nt] = vzero;

  const __bf16* arow = A + (size_t)((mtile << 4) + (lane & 15)) * K;
#pragma unroll
  for (int kt = 0; kt < KT; ++kt) {
    if (kt + 1 < KT) __builtin_prefetch(arow + (kt + 1) * 32, 0, 3);
    v8bf a0 = *(const v8bf*)(arow + kt * 32 + khalf * 16);
    v8bf a1 = *(const v8bf*)(arow + kt * 32 + khalf * 16 + 8);
    v16bf af;
#pragma unroll
    for (int t = 0; t < 8; ++t) { af[t] = a0[t]; af[t + 8] = a1[t]; }
#pragma unroll
    for (int nt = 0; nt < NT; ++nt) {
      v16bf bfr = *(const v16bf*)(&sW[((nt * KT + kt) * 32 + lane) * 16]);
      acc[nt] = __builtin_amdgcn_wmma_f32_16x16x32_bf16(
          false, af, false, bfr, (short)0, acc[nt], false, false);
    }
  }

#pragma unroll
  for (int nt = 0; nt < NT; ++nt) {
    int n = nt * 16 + (lane & 15);
    float bv = bias[n];
#pragma unroll
    for (int i = 0; i < 8; ++i) {
      int m_out = (mtile << 4) + (khalf << 3) + i;
      float v = acc[nt][i] + bv;
      if (ACT == 1) v = fmaxf(v, 0.0f);
      else if (ACT == 2) v = geluf(v);
      if (STORE == 0) {
        ((float*)Out)[(size_t)m_out * N + n] = v;
      } else if (STORE == 1) {
        ((__bf16*)Out)[(size_t)m_out * N + n] = f2bf(v);
      } else if (STORE == 2) {  // row m = b*1024 + hw  ->  (B, N, 32, 32)
        int b = m_out >> 10, hw = m_out & 1023;
        ((float*)Out)[((size_t)b * N + n) * 1024 + hw] = v;
      } else {
        size_t idx = (size_t)m_out * N + n;
        ((float*)Out)[idx] = resid[idx] + v;
      }
    }
  }
}

// ---------------------------------------------------------------------------
// Depthwise 3x3 conv (pad 1) + bias + GELU; out bf16 (BN, HID) row-major.
// ---------------------------------------------------------------------------
__global__ void __launch_bounds__(1024) dwconv_gelu(
    const float* __restrict__ Hc, const float* __restrict__ wd,
    const float* __restrict__ bd, __bf16* __restrict__ out) {
  const int tx = threadIdx.x, ty = threadIdx.y;
  const int p = blockIdx.x;             // b*HID + ch
  const int ch = p & (HID_ - 1), b = p >> 8;
  const size_t base = (size_t)p * 1024;
  float s = 0.f;
#pragma unroll
  for (int dy = 0; dy < 3; ++dy) {
#pragma unroll
    for (int dx = 0; dx < 3; ++dx) {
      int yy = ty + dy - 1, xx = tx + dx - 1;
      if (yy >= 0 && yy < 32 && xx >= 0 && xx < 32)
        s += wd[ch * 9 + dy * 3 + dx] * Hc[base + yy * 32 + xx];
    }
  }
  s = geluf(s + bd[ch]);
  out[((size_t)b * 1024 + ty * 32 + tx) * HID_ + ch] = f2bf(s);
}

// ===========================================================================
extern "C" void kernel_launch(void* const* d_in, const int* in_sizes, int n_in,
                              void* d_out, int out_size, void* d_ws, size_t ws_size,
                              hipStream_t stream) {
  const float* x      = (const float*)d_in[0];
  const float* ln1_g  = (const float*)d_in[1];
  const float* ln1_b  = (const float*)d_in[2];
  const float* w_ll   = (const float*)d_in[3];
  const float* w1     = (const float*)d_in[4];
  const float* w2     = (const float*)d_in[5];
  const float* b1     = (const float*)d_in[6];
  const float* b2     = (const float*)d_in[7];
  const float* ln2_g  = (const float*)d_in[8];
  const float* ln2_b  = (const float*)d_in[9];
  const float* lin1_w = (const float*)d_in[10];
  const float* lin1_b = (const float*)d_in[11];
  const float* dw_w   = (const float*)d_in[12];
  const float* dw_b   = (const float*)d_in[13];
  const float* lin2_w = (const float*)d_in[14];
  const float* lin2_b = (const float*)d_in[15];
  float* out = (float*)d_out;

  char* ws = (char*)d_ws;
  const size_t MB = 1ull << 20;
  // Time-sliced arena (offsets in bytes); total footprint ~833 MB.
  float*  xi    = (float*)(ws + 0);          // [0,64M)    also lh2, then xb
  float*  lo    = (float*)(ws + 64 * MB);    // [64,128M)  also XRI, then hl2
  float*  hi    = (float*)(ws + 128 * MB);   // [128,192M) also hh2
  float*  ll    = (float*)(ws + 192 * MB);
  float*  lh    = (float*)(ws + 256 * MB);   // also H1, then lo2
  float*  hl    = (float*)(ws + 320 * MB);   // also hi2
  float*  hh    = (float*)(ws + 384 * MB);   // also y1
  float*  H2    = (float*)(ws + 448 * MB);   // 192MB; later Hc (256MB)
  __bf16* conv  = (__bf16*)(ws + 704 * MB);  // 128MB
  // small weight area
  float*  W1big = (float*)(ws + 832 * MB);
  float*  W2big = W1big + 128 * 128;
  float*  b1big = W2big + 128 * 128;
  float*  b2big = b1big + 128;
  __bf16* F1    = (__bf16*)(b2big + 128);
  __bf16* F2    = F1 + 128 * 128;
  __bf16* Flin1 = F2 + 128 * 128;
  __bf16* Flin2 = Flin1 + 64 * 256;

  __bf16* XRI  = (__bf16*)lo;   // (R,128) bf16 ~100MB, spans lo+hi regions
  __bf16* H1   = (__bf16*)lh;   // (R,128) bf16, spans lh+hl regions
  float*  lh2  = xi;            // reuse after forward pass
  float*  hl2  = lo;
  float*  hh2  = hi;
  float*  lo2  = lh;
  float*  hi2  = hl;
  float*  y1   = hh;
  float*  Hc   = H2;            // (B,HID,32,32) f32, 256MB
  __bf16* xb   = (__bf16*)xi;   // LN2 output (BN,64) bf16

  dim3 plane(32, 32);

  // -- weight prep ----------------------------------------------------------
  build_wbig<<<64, 256, 0, stream>>>(w1, b1, W1big, b1big);
  build_wbig<<<64, 256, 0, stream>>>(w2, b2, W2big, b2big);
  prep_wfrag<<<64, 256, 0, stream>>>(W1big, F1, 128, 128);
  prep_wfrag<<<64, 256, 0, stream>>>(W2big, F2, 128, 128);
  prep_wfrag<<<64, 256, 0, stream>>>(lin1_w, Flin1, 64, 256);
  prep_wfrag<<<64, 256, 0, stream>>>(lin2_w, Flin2, 256, 64);

  // -- SVT branch -----------------------------------------------------------
  ln1_to_nchw<<<BN / 4, 128, 0, stream>>>(x, ln1_g, ln1_b, xi);
  fwd_row_filter<<<BB * CCH, plane, 0, stream>>>(xi, lo, hi);
  fwd_col_filter<<<BB * CCH, plane, 0, stream>>>(lo, hi, ll, lh, hl, hh);
  ll_scale<<<S1 / 256, 256, 0, stream>>>(ll, w_ll);
  q2c_pack<<<BB * 256, 64, 0, stream>>>(lh, hl, hh, XRI);

  gemm_bf16_wmma<128, 128, 1, 1><<<RROWS / 128, 256, 0, stream>>>(
      XRI, F1, b1big, (void*)H1, nullptr);
  gemm_bf16_wmma<128, 128, 0, 0><<<RROWS / 128, 256, 0, stream>>>(
      H1, F2, b2big, (void*)H2, nullptr);

  c2q_unpack<<<BB * 256, 64, 0, stream>>>(H2, lh2, hl2, hh2);
  inv_col_filter<<<BB * CCH, plane, 0, stream>>>(ll, lh2, hl2, hh2, lo2, hi2);
  inv_row_resid<<<BB * CCH, plane, 0, stream>>>(lo2, hi2, x, y1);

  // -- LeFF branch ----------------------------------------------------------
  ln2_to_bf16<<<BN / 4, 128, 0, stream>>>(y1, ln2_g, ln2_b, xb);
  gemm_bf16_wmma<64, 256, 2, 2><<<BN / 128, 256, 0, stream>>>(
      xb, Flin1, lin1_b, (void*)Hc, nullptr);
  dwconv_gelu<<<BB * HID_, plane, 0, stream>>>(Hc, dw_w, dw_b, conv);
  gemm_bf16_wmma<256, 64, 0, 3><<<BN / 128, 256, 0, stream>>>(
      conv, Flin2, lin2_b, (void*)out, y1);
}
